// QKVAttention_71030169141624
// MI455X (gfx1250) — compile-verified, matrix operand order checked by reading, not run on hardware
//
#include <hip/hip_runtime.h>
#include <hip/hip_bf16.h>

typedef _Float16 v16h __attribute__((ext_vector_type(16)));
typedef _Float16 v8h  __attribute__((ext_vector_type(8)));
typedef _Float16 v4h  __attribute__((ext_vector_type(4)));
typedef float    v8f  __attribute__((ext_vector_type(8)));

#define L        1024
#define CH       64
#define BLOCK_M  64
#define BLOCK_N  64
#define NWAVES   4
#define KPITCH   72   // halfs per kt row: [s 64][c 64 + pad]
#define VPITCH   72   // halfs per vt row: [c 64][s 64 + pad]
#define PPITCH   72   // halfs per pt row: [t 16][s 64 + pad] (per wave)
#define OPITCH   68   // floats per ot row: [c 64][t 64 + pad]

// Build a v16h A/B fragment from two contiguous 8-half chunks
#define CAT16(lo, hi) __builtin_shufflevector(lo, hi, 0,1,2,3,4,5,6,7,8,9,10,11,12,13,14,15)

#define WMMA_F16(A, B, C) \
    __builtin_amdgcn_wmma_f32_16x16x32_f16(false, (A), false, (B), (short)0, (C), false, false)

__global__ __launch_bounds__(128)
void qkv_attn_fa(const float* __restrict__ qkv, float* __restrict__ out) {
    __shared__ alignas(16) _Float16 kt[BLOCK_N * KPITCH];          // [s][c]  c contiguous
    __shared__ alignas(16) _Float16 vt[CH * VPITCH];               // [c][s]  s contiguous
    __shared__ alignas(16) _Float16 pt[NWAVES * 16 * PPITCH];      // per-wave [t][s]
    __shared__ alignas(16) float    ot[CH * OPITCH];               // [c][t]  output staging

    const int bh   = blockIdx.x;          // 0..127  (b*16 + h)
    const int tblk = blockIdx.y;          // 0..15
    const int b    = bh >> 4;
    const int h    = bh & 15;
    const int tid  = threadIdx.x;
    const int wave = tid >> 5;
    const int lane = tid & 31;
    const int half = lane >> 4;
    const int l16  = lane & 15;

    const size_t base = (size_t)b * 3072 * L + (size_t)h * CH * L;
    const float* __restrict__ q = qkv + base;
    const float* __restrict__ k = qkv + base + (size_t)1024 * L;
    const float* __restrict__ v = qkv + base + (size_t)2048 * L;

    const int t0 = tblk * BLOCK_M + wave * 16;   // this wave's first query row

    // ---- Q A-fragments (16t x 32c each) ----
    // scale^2 = 1/sqrt(ch) AND log2(e) folded into Q so the loop uses bare v_exp_f32:
    //   exp(q.k * 0.125) == exp2((q * 0.125*log2e) . k)
    // A layout: lane holds row M=l16, element e -> K = (e&7) + ((e>>3)<<4) + (half<<3)
    const float qscale = 0.125f * 1.44269504088896340736f;
    v16h aq0, aq1;
    {
        const int t = t0 + l16;
        #pragma unroll
        for (int e = 0; e < 16; ++e) {
            const int c0 = (e & 7) + ((e >> 3) << 4) + (half << 3);
            aq0[e] = (_Float16)(q[(size_t)c0        * L + t] * qscale);
            aq1[e] = (_Float16)(q[(size_t)(c0 + 32) * L + t] * qscale);
        }
    }

    v8f acc[4] = {v8f{}, v8f{}, v8f{}, v8f{}};
    v8f lrow = {};   // per-lane partial row sums (C layout), reduced once at the end

    _Float16* __restrict__ ptw = pt + wave * 16 * PPITCH;
    const int sb = half << 3;

    for (int s0 = 0; s0 < L; s0 += BLOCK_N) {
        // ---- cooperative K/V tile load -> LDS (f16), float4 coalesced along s ----
        __syncthreads();
        #pragma unroll
        for (int j = 0; j < 8; ++j) {
            const int linear = j * 512 + tid * 4;          // 4096 elements per tile
            const int s = linear & 63;
            const int c = linear >> 6;
            const float4 kv_k = *(const float4*)(k + (size_t)c * L + s0 + s);
            const float4 kv_v = *(const float4*)(v + (size_t)c * L + s0 + s);
            kt[(s + 0) * KPITCH + c] = (_Float16)kv_k.x;
            kt[(s + 1) * KPITCH + c] = (_Float16)kv_k.y;
            kt[(s + 2) * KPITCH + c] = (_Float16)kv_k.z;
            kt[(s + 3) * KPITCH + c] = (_Float16)kv_k.w;
            *(v4h*)&vt[c * VPITCH + s] =
                (v4h){(_Float16)kv_v.x, (_Float16)kv_v.y, (_Float16)kv_v.z, (_Float16)kv_v.w};
        }
        // prefetch next K/V tile rows (-> global_prefetch_b8)
        if (s0 + BLOCK_N < L) {
            const int c2 = tid & 63;
            const float* pf = (tid < 64 ? k : v) + (size_t)c2 * L + s0 + BLOCK_N;
            __builtin_prefetch(pf, 0, 1);
        }
        __syncthreads();

        // ---- S = (Q^T K) tile: [16t x 64s] as four 16x16 C tiles ----
        v8f st[4];
        #pragma unroll
        for (int ss = 0; ss < 4; ++ss) {
            const _Float16* kr = kt + (ss * 16 + l16) * KPITCH;
            v16h bk0 = CAT16(*(const v8h*)(kr + sb),      *(const v8h*)(kr + sb + 16));
            v16h bk1 = CAT16(*(const v8h*)(kr + sb + 32), *(const v8h*)(kr + sb + 48));
            v8f cz = {};
            cz = WMMA_F16(aq0, bk0, cz);
            cz = WMMA_F16(aq1, bk1, cz);
            st[ss] = cz;
        }

        // ---- P = exp2(S) (log2e pre-folded); per-lane partial row sums, no shuffles ----
        #pragma unroll
        for (int ss = 0; ss < 4; ++ss) {
            #pragma unroll
            for (int i = 0; i < 8; ++i) {
                const float e = __builtin_amdgcn_exp2f(st[ss][i]);   // bare v_exp_f32
                st[ss][i] = e;
                lrow[i] += e;
            }
        }

        // ---- transpose P (C layout -> A layout) through per-wave LDS ----
        #pragma unroll
        for (int ss = 0; ss < 4; ++ss) {
            #pragma unroll
            for (int i = 0; i < 8; ++i) {
                ptw[(i + (half << 3)) * PPITCH + ss * 16 + l16] = (_Float16)st[ss][i];
            }
        }
        asm volatile("s_wait_dscnt 0" ::: "memory");   // cross-lane same-wave RAW on LDS

        const _Float16* pr = ptw + l16 * PPITCH;
        v16h pA0 = CAT16(*(const v8h*)(pr + sb),      *(const v8h*)(pr + sb + 16));
        v16h pA1 = CAT16(*(const v8h*)(pr + sb + 32), *(const v8h*)(pr + sb + 48));

        // ---- O[t,c] += P[t,s] * V^T[s,c] : 4 c-tiles, K=64 in two WMMAs each ----
        #pragma unroll
        for (int ct = 0; ct < 4; ++ct) {
            const _Float16* vr = vt + (ct * 16 + l16) * VPITCH;
            v16h bv0 = CAT16(*(const v8h*)(vr + sb),      *(const v8h*)(vr + sb + 16));
            v16h bv1 = CAT16(*(const v8h*)(vr + sb + 32), *(const v8h*)(vr + sb + 48));
            acc[ct] = WMMA_F16(pA0, bv0, acc[ct]);
            acc[ct] = WMMA_F16(pA1, bv1, acc[ct]);
        }
    }

    // ---- one-time row-sum reduction across the 16 lanes holding each row ----
    #pragma unroll
    for (int off = 1; off < 16; off <<= 1) {
        #pragma unroll
        for (int i = 0; i < 8; ++i) lrow[i] += __shfl_xor(lrow[i], off, 32);
    }
    v8f rinv;
    #pragma unroll
    for (int i = 0; i < 8; ++i) rinv[i] = 1.0f / lrow[i];

    // ---- stage normalized O to LDS [c][t], then coalesced global store ----
    #pragma unroll
    for (int ct = 0; ct < 4; ++ct) {
        #pragma unroll
        for (int i = 0; i < 8; ++i) {
            ot[(ct * 16 + l16) * OPITCH + wave * 16 + i + (half << 3)] = acc[ct][i] * rinv[i];
        }
    }
    __syncthreads();

    float* __restrict__ ob = out + (size_t)bh * CH * L + (size_t)tblk * BLOCK_M;
    #pragma unroll
    for (int idx = tid; idx < CH * BLOCK_M; idx += 128) {
        const int t = idx & 63;
        const int c = idx >> 6;
        ob[(size_t)c * L + t] = ot[c * OPITCH + t];
    }
}

extern "C" void kernel_launch(void* const* d_in, const int* in_sizes, int n_in,
                              void* d_out, int out_size, void* d_ws, size_t ws_size,
                              hipStream_t stream) {
    const float* qkv = (const float*)d_in[0];
    float* out = (float*)d_out;
    (void)in_sizes; (void)n_in; (void)out_size; (void)d_ws; (void)ws_size;
    dim3 grid(128, 16);   // 128 (b,h) pairs x 16 query-row blocks of 64
    qkv_attn_fa<<<grid, dim3(128), 0, stream>>>(qkv, out);
}
